// RoIHeadsNew_24378234372504
// MI455X (gfx1250) — matrix operand order repaired and checked by reading, not compiled, not run on hardware
//
#include <hip/hip_runtime.h>
#include <stdint.h>

// ---------------------------------------------------------------------------
// RoI keypoint head for MI455X (gfx1250, wave32, WMMA).
//
// Per image (n,k): out(112x112) = W(112x56) @ img(56x56) @ W^T(56x112),
// W = fixed bicubic interpolation matrix (56 -> 112, A = -0.75, same for x/y),
// then argmax(out) -> (x, y, 1, score) through the ROI box.
//
// Both GEMMs run on v_wmma_f32_16x16x32_f16 (f16 in, f32 accumulate).
// K padded 56 -> 64 (2 k-slabs of 32), M = 112 (7 tiles), N = 64 / 112.
// One 256-thread block (8 waves) per image; all intermediates in LDS.
// Two output tiles are processed per wave iteration with interleaved WMMAs
// so the WMMA->WMMA accumulator hazard (1 NOP + 4 coexec) is hidden by the
// sibling tile's instructions instead of v_nops.
// ---------------------------------------------------------------------------

typedef _Float16 v8h  __attribute__((ext_vector_type(8)));
typedef _Float16 v16h __attribute__((ext_vector_type(16)));
typedef float    v8f  __attribute__((ext_vector_type(8)));
typedef float    f4   __attribute__((ext_vector_type(4)));

#define KCH        17
#define IN_HW      56
#define OUT_HW     112
#define A_CUBIC    (-0.75f)
#define W_PITCH    64   // halves per row of weight matrix (112 x 64, row-major)
#define IMG_PITCH  72   // halves per column of imgT (64 cols x 72)
#define TMP_PITCH  72   // halves per row of tmp (112 x 72)

// A-operand (16x32 f16): lane m = L%16, khalf = L/16.
// v16h elems 0..7 = A[m][k0..k0+7], elems 8..15 = A[m][k0+16..k0+23],
// k0 = 32*kslab + 8*khalf.  Row-major storage -> two ds_load_b128.
__device__ __forceinline__ v16h load_a_tile(const _Float16* base, int pitch,
                                            int m0, int kslab, int lane) {
  int m     = m0 + (lane & 15);
  int khalf = lane >> 4;
  const _Float16* p = base + m * pitch + kslab * 32 + khalf * 8;
  v8h lo = *(const v8h*)p;
  v8h hi = *(const v8h*)(p + 16);
  return __builtin_shufflevector(lo, hi, 0, 1, 2, 3, 4, 5, 6, 7,
                                         8, 9, 10, 11, 12, 13, 14, 15);
}

// B-operand (32x16 f16): lane n = L%16, khalf = L/16.
// v16h elems 0..7 = B[k0..k0+7][n], elems 8..15 = B[k0+16..k0+23][n], k0 = 32*kslab + 8*khalf.
// k-contiguous (transposed) storage -> two ds_load_b128.
__device__ __forceinline__ v16h load_b_tile(const _Float16* base, int pitch,
                                            int n0, int kslab, int lane) {
  int n     = n0 + (lane & 15);
  int khalf = lane >> 4;
  const _Float16* p = base + n * pitch + kslab * 32 + khalf * 8;
  v8h lo = *(const v8h*)p;
  v8h hi = *(const v8h*)(p + 16);
  return __builtin_shufflevector(lo, hi, 0, 1, 2, 3, 4, 5, 6, 7,
                                         8, 9, 10, 11, 12, 13, 14, 15);
}

// D layout: elem r -> (M = 8*(lane/16)+r, N = lane%16); store tmp tile as f16.
__device__ __forceinline__ void store_tmp_tile(_Float16* sTmp, const v8f& acc,
                                               int mt, int nt, int lane) {
  int n     = nt * 16 + (lane & 15);
  int mbase = mt * 16 + ((lane >> 4) << 3);
  #pragma unroll
  for (int r = 0; r < 8; ++r)
    sTmp[(mbase + r) * TMP_PITCH + n] = (_Float16)acc[r];
}

// Monotone float->u32 key | (12543-pos): max key == argmax with first-
// occurrence tie-break (matches jnp.argmax).
__device__ __forceinline__ void argmax_update(const v8f& acc, int mt, int nt,
                                              int lane,
                                              unsigned long long& bestKey) {
  int w     = nt * 16 + (lane & 15);
  int hbase = mt * 16 + ((lane >> 4) << 3);
  #pragma unroll
  for (int r = 0; r < 8; ++r) {
    int pos = (hbase + r) * OUT_HW + w;
    uint32_t u = __float_as_uint(acc[r]);
    u = (u & 0x80000000u) ? ~u : (u | 0x80000000u);
    unsigned long long key =
        ((unsigned long long)u << 32) | (uint32_t)(12543 - pos);
    bestKey = key > bestKey ? key : bestKey;
  }
}

__device__ __forceinline__ unsigned long long shfl_xor_u64(unsigned long long v,
                                                           int mask) {
  int lo = __shfl_xor((int)(unsigned)(v & 0xFFFFFFFFull), mask, 32);
  int hi = __shfl_xor((int)(unsigned)(v >> 32), mask, 32);
  return ((unsigned long long)(unsigned)hi << 32) | (unsigned)lo;
}

__global__ __launch_bounds__(256) void roi_keypoint_kernel(
    const float* __restrict__ maps, const float* __restrict__ rois,
    float* __restrict__ out) {
  __shared__ alignas(32) _Float16 sW[OUT_HW * W_PITCH];      // 14336 B
  __shared__ alignas(32) _Float16 sImgT[64 * IMG_PITCH];     //  9216 B
  __shared__ alignas(32) _Float16 sTmp[OUT_HW * TMP_PITCH];  // 16128 B
  __shared__ unsigned long long sKey;

  const int tid  = threadIdx.x;
  const int lane = tid & 31;
  // Wave id as an SGPR: loop control below becomes scalar branches, which
  // also trivially guarantees EXEC == all-ones around every WMMA.
  const int wv   = __builtin_amdgcn_readfirstlane(tid >> 5);

  if (tid == 0) sKey = 0ull;

  // ---- zero weight matrix + imgT (incl. K/N padding) ----------------------
  uint32_t* wz = (uint32_t*)sW;
  for (int i = tid; i < OUT_HW * (W_PITCH / 2); i += 256) wz[i] = 0u;
  uint32_t* iz = (uint32_t*)sImgT;
  for (int i = tid; i < 64 * (IMG_PITCH / 2); i += 256) iz[i] = 0u;
  __syncthreads();

  // ---- bicubic interpolation matrix (identical for x and y axes) ----------
  if (tid < OUT_HW) {
    float src = (tid + 0.5f) * 0.5f - 0.5f;   // scale = 56/112 = 0.5
    float i0  = floorf(src);
    float t   = src - i0;
    float s   = 1.0f - t;
    float w[4];
    w[0] = A_CUBIC * (t * t * t - 2.0f * t * t + t);
    w[1] = (A_CUBIC + 2.0f) * t * t * t - (A_CUBIC + 3.0f) * t * t + 1.0f;
    w[2] = (A_CUBIC + 2.0f) * s * s * s - (A_CUBIC + 3.0f) * s * s + 1.0f;
    w[3] = A_CUBIC * (s * s * s - 2.0f * s * s + s);
    int ibase = (int)i0 - 1;
    _Float16* row = sW + tid * W_PITCH;
    #pragma unroll
    for (int j = 0; j < 4; ++j) {
      int idx = min(IN_HW - 1, max(0, ibase + j));   // border clamp collides
      row[idx] = (_Float16)((float)row[idx] + w[j]); // row owned by one thread
    }
  }

  // ---- load image as float4 (global_load_b128), f32->f16, transpose -------
  // 56x56 floats = 784 float4 exactly (14 per row).
  const f4* img4 = (const f4*)(maps + (size_t)blockIdx.x * (IN_HW * IN_HW));
  for (int i = tid; i < 784; i += 256) {
    f4 v = img4[i];
    int y  = i / 14;            // source row
    int x0 = (i - y * 14) * 4;  // source col of first element
    #pragma unroll
    for (int j = 0; j < 4; ++j)
      sImgT[(x0 + j) * IMG_PITCH + y] = (_Float16)v[j];
  }
  __syncthreads();

  // ---- GEMM1: tmp(112x64) = W @ img   [7x4 = 28 tiles, paired] ------------
  // Pair (t, t+8): same ntile (B shared), mtiles differ by 2; interleaved
  // WMMAs hide the accumulator RAW hazard.
  {
    int t = wv;
    for (; t + 8 < 28; t += 16) {
      int nt = t & 3, mt0 = t >> 2, mt1 = mt0 + 2;
      v16h b0  = load_b_tile(sImgT, IMG_PITCH, nt * 16, 0, lane);
      v16h b1  = load_b_tile(sImgT, IMG_PITCH, nt * 16, 1, lane);
      v16h a00 = load_a_tile(sW, W_PITCH, mt0 * 16, 0, lane);
      v16h a01 = load_a_tile(sW, W_PITCH, mt0 * 16, 1, lane);
      v16h a10 = load_a_tile(sW, W_PITCH, mt1 * 16, 0, lane);
      v16h a11 = load_a_tile(sW, W_PITCH, mt1 * 16, 1, lane);
      v8f acc0 = {}, acc1 = {};
      acc0 = __builtin_amdgcn_wmma_f32_16x16x32_f16(false, a00, false, b0,
                                                    (short)0, acc0, false, false);
      acc1 = __builtin_amdgcn_wmma_f32_16x16x32_f16(false, a10, false, b0,
                                                    (short)0, acc1, false, false);
      acc0 = __builtin_amdgcn_wmma_f32_16x16x32_f16(false, a01, false, b1,
                                                    (short)0, acc0, false, false);
      acc1 = __builtin_amdgcn_wmma_f32_16x16x32_f16(false, a11, false, b1,
                                                    (short)0, acc1, false, false);
      store_tmp_tile(sTmp, acc0, mt0, nt, lane);
      store_tmp_tile(sTmp, acc1, mt1, nt, lane);
    }
    for (; t < 28; t += 8) {   // remainder (waves 4..7: one single tile)
      int nt = t & 3, mt = t >> 2;
      v8f acc = {};
      #pragma unroll
      for (int ks = 0; ks < 2; ++ks) {
        v16h a = load_a_tile(sW, W_PITCH, mt * 16, ks, lane);
        v16h b = load_b_tile(sImgT, IMG_PITCH, nt * 16, ks, lane);
        acc = __builtin_amdgcn_wmma_f32_16x16x32_f16(false, a, false, b,
                                                     (short)0, acc, false, false);
      }
      store_tmp_tile(sTmp, acc, mt, nt, lane);
    }
  }
  __syncthreads();

  // ---- GEMM2: out(112x112) = tmp @ W^T  [7x7 = 49 tiles, paired] + argmax -
  unsigned long long bestKey = 0ull;
  {
    int t = wv;
    for (; t + 8 < 49; t += 16) {
      int mt0 = t / 7, nt0 = t - mt0 * 7;
      int t1 = t + 8;
      int mt1 = t1 / 7, nt1 = t1 - mt1 * 7;
      v16h a00 = load_a_tile(sTmp, TMP_PITCH, mt0 * 16, 0, lane);
      v16h a01 = load_a_tile(sTmp, TMP_PITCH, mt0 * 16, 1, lane);
      v16h b00 = load_b_tile(sW, W_PITCH, nt0 * 16, 0, lane);
      v16h b01 = load_b_tile(sW, W_PITCH, nt0 * 16, 1, lane);
      v16h a10 = load_a_tile(sTmp, TMP_PITCH, mt1 * 16, 0, lane);
      v16h a11 = load_a_tile(sTmp, TMP_PITCH, mt1 * 16, 1, lane);
      v16h b10 = load_b_tile(sW, W_PITCH, nt1 * 16, 0, lane);
      v16h b11 = load_b_tile(sW, W_PITCH, nt1 * 16, 1, lane);
      v8f acc0 = {}, acc1 = {};
      acc0 = __builtin_amdgcn_wmma_f32_16x16x32_f16(false, a00, false, b00,
                                                    (short)0, acc0, false, false);
      acc1 = __builtin_amdgcn_wmma_f32_16x16x32_f16(false, a10, false, b10,
                                                    (short)0, acc1, false, false);
      acc0 = __builtin_amdgcn_wmma_f32_16x16x32_f16(false, a01, false, b01,
                                                    (short)0, acc0, false, false);
      acc1 = __builtin_amdgcn_wmma_f32_16x16x32_f16(false, a11, false, b11,
                                                    (short)0, acc1, false, false);
      argmax_update(acc0, mt0, nt0, lane, bestKey);
      argmax_update(acc1, mt1, nt1, lane, bestKey);
    }
    for (; t < 49; t += 8) {   // remainder (wave 0: tile 48)
      int mt = t / 7, nt = t - mt * 7;
      v8f acc = {};
      #pragma unroll
      for (int ks = 0; ks < 2; ++ks) {
        v16h a = load_a_tile(sTmp, TMP_PITCH, mt * 16, ks, lane);
        v16h b = load_b_tile(sW, W_PITCH, nt * 16, ks, lane);
        acc = __builtin_amdgcn_wmma_f32_16x16x32_f16(false, a, false, b,
                                                     (short)0, acc, false, false);
      }
      argmax_update(acc, mt, nt, lane, bestKey);
    }
  }

  // ---- wave reduce, then one ds_max_u64 per wave --------------------------
  #pragma unroll
  for (int off = 16; off > 0; off >>= 1) {
    unsigned long long o = shfl_xor_u64(bestKey, off);
    bestKey = o > bestKey ? o : bestKey;
  }
  if (lane == 0) atomicMax(&sKey, bestKey);
  __syncthreads();

  // ---- decode argmax, map through ROI box, emit (x, y, 1, score) ----------
  if (tid == 0) {
    unsigned long long key = sKey;
    int pos = 12543 - (int)(key & 0xFFFFFFFFull);
    uint32_t mono  = (uint32_t)(key >> 32);
    uint32_t ubits = (mono & 0x80000000u) ? (mono & 0x7FFFFFFFu) : ~mono;
    float score = __uint_as_float(ubits);
    int xi = pos % OUT_HW, yi = pos / OUT_HW;
    int n = blockIdx.x / KCH;
    float r0 = rois[n * 4 + 0], r1 = rois[n * 4 + 1];
    float r2 = rois[n * 4 + 2], r3 = rois[n * 4 + 3];
    float wcorr = fmaxf(r2 - r0, 1.0f) * (1.0f / (float)OUT_HW);
    float hcorr = fmaxf(r3 - r1, 1.0f) * (1.0f / (float)OUT_HW);
    float* o = out + (size_t)blockIdx.x * 4;
    o[0] = ((float)xi + 0.5f) * wcorr + r0;
    o[1] = ((float)yi + 0.5f) * hcorr + r1;
    o[2] = 1.0f;
    o[3] = score;
  }
}

extern "C" void kernel_launch(void* const* d_in, const int* in_sizes, int n_in,
                              void* d_out, int out_size, void* d_ws, size_t ws_size,
                              hipStream_t stream) {
  const float* maps = (const float*)d_in[0];   // (512, 17, 56, 56) f32
  const float* rois = (const float*)d_in[1];   // (512, 4) f32
  float* out = (float*)d_out;                  // (512, 17, 4) f32
  int nimg = in_sizes[0] / (IN_HW * IN_HW);    // N*K = 8704 images
  roi_keypoint_kernel<<<nimg, 256, 0, stream>>>(maps, rois, out);
}
